// Quantize_No_Transformer_30683246363108
// MI455X (gfx1250) — matrix operand order repaired and checked by reading, not compiled, not run on hardware
//
#include <hip/hip_runtime.h>
#include <cstdint>
#include <cstddef>

#define Bn 16
#define Cc 512
#define Hh 16
#define Dd 64
#define NE 1024
#define NT (Hh * Dd)              // 1024 tokens
#define BC (Bn * Cc)              // 8192
#define NROWS (Bn * Cc * Hh)      // 131072 VQ rows
#define OUT_ELEMS (Bn * Cc * Hh * Dd) // 8388608

typedef __attribute__((ext_vector_type(16))) __bf16 v16bf_t;
typedef __attribute__((ext_vector_type(8)))  float  v8f_t;
typedef __attribute__((ext_vector_type(8)))  unsigned int v8u_t;

static __device__ __forceinline__ unsigned short f2bf(float f) {
    unsigned u = __builtin_bit_cast(unsigned, f);
    u += 0x7FFFu + ((u >> 16) & 1u);           // round-to-nearest-even
    return (unsigned short)(u >> 16);
}

// A-fragment (16x32 bf16, M x K): lane l, elem e -> M=l&15, K=(e>>3)*16+(l>>4)*8+(e&7)
static __device__ __forceinline__ v16bf_t frag_a(const unsigned short* base, int rowStride,
                                                 int mr, int kb, int lane) {
    const int half = lane >> 4, rl = lane & 15;
    const unsigned short* p = base + (size_t)(mr + rl) * rowStride + kb + half * 8;
    uint4 c0 = *reinterpret_cast<const uint4*>(p);
    uint4 c1 = *reinterpret_cast<const uint4*>(p + 16);
    v8u_t pk;
    pk[0] = c0.x; pk[1] = c0.y; pk[2] = c0.z; pk[3] = c0.w;
    pk[4] = c1.x; pk[5] = c1.y; pk[6] = c1.z; pk[7] = c1.w;
    return __builtin_bit_cast(v16bf_t, pk);
}

// B-fragment (32x16 bf16, K x N) from N-major storage Bt[n][k]: lane l, elem e -> N=l&15, K=(e>>3)*16+(l>>4)*8+(e&7)
static __device__ __forceinline__ v16bf_t frag_b(const unsigned short* bt, int kStride,
                                                 int nb, int kb, int lane) {
    const int half = lane >> 4, nl = lane & 15;
    const unsigned short* p = bt + (size_t)(nb + nl) * kStride + kb + half * 8;
    uint4 c0 = *reinterpret_cast<const uint4*>(p);
    uint4 c1 = *reinterpret_cast<const uint4*>(p + 16);
    v8u_t pk;
    pk[0] = c0.x; pk[1] = c0.y; pk[2] = c0.z; pk[3] = c0.w;
    pk[4] = c1.x; pk[5] = c1.y; pk[6] = c1.z; pk[7] = c1.w;
    return __builtin_bit_cast(v16bf_t, pk);
}

static __device__ __forceinline__ v8f_t wmma_bf16(v16bf_t a, v16bf_t b, v8f_t c) {
    return __builtin_amdgcn_wmma_f32_16x16x32_bf16(false, a, false, b, (short)0, c, false, false);
}

static __device__ __forceinline__ v8f_t zero8() {
    v8f_t z; for (int i = 0; i < 8; ++i) z[i] = 0.f; return z;
}

static __device__ __forceinline__ uint4 pack8bf(const float* v) {
    uint4 o;
    o.x = (unsigned)f2bf(v[0]) | ((unsigned)f2bf(v[1]) << 16);
    o.y = (unsigned)f2bf(v[2]) | ((unsigned)f2bf(v[3]) << 16);
    o.z = (unsigned)f2bf(v[4]) | ((unsigned)f2bf(v[5]) << 16);
    o.w = (unsigned)f2bf(v[6]) | ((unsigned)f2bf(v[7]) << 16);
    return o;
}

// ---------------------------------------------------------------- K0: prep
__global__ void k_prep(const float* __restrict__ W_lin, const float* __restrict__ embed,
                       unsigned short* __restrict__ W_t, unsigned short* __restrict__ ebt,
                       float* __restrict__ halfnorm) {
    int gid = blockIdx.x * 256 + threadIdx.x;
    if (gid < Cc * Cc) {            // W_t[n][k] = W_lin[k][n]
        int n = gid / Cc, k = gid % Cc;
        W_t[(size_t)n * Cc + k] = f2bf(W_lin[(size_t)k * Cc + n]);
    }
    if (gid < NE * Dd) {            // ebt[n][k] = embed[k][n]
        int n = gid / Dd, k = gid % Dd;
        ebt[(size_t)n * Dd + k] = f2bf(embed[(size_t)k * NE + n]);
    }
    if (gid < NE) {                 // halfnorm[n] = 0.5 * sum_k embed[k][n]^2
        float s = 0.f;
        for (int k = 0; k < Dd; ++k) { float e = embed[(size_t)k * NE + gid]; s += e * e; }
        halfnorm[gid] = 0.5f * s;
    }
}

// ---------------------------------------------------------------- K1: per-(b,c) stats
__global__ void k_stats(const float* __restrict__ cF, const float* __restrict__ sF,
                        float* __restrict__ stats) {
    __shared__ float red[4][256];
    int bc = blockIdx.x, tid = threadIdx.x;
    float4 x = reinterpret_cast<const float4*>(cF + (size_t)bc * 1024)[tid];
    float4 y = reinterpret_cast<const float4*>(sF + (size_t)bc * 1024)[tid];
    red[0][tid] = x.x + x.y + x.z + x.w;
    red[1][tid] = x.x * x.x + x.y * x.y + x.z * x.z + x.w * x.w;
    red[2][tid] = y.x + y.y + y.z + y.w;
    red[3][tid] = y.x * y.x + y.y * y.y + y.z * y.z + y.w * y.w;
    __syncthreads();
    for (int o = 128; o > 0; o >>= 1) {
        if (tid < o)
            for (int j = 0; j < 4; ++j) red[j][tid] += red[j][tid + o];
        __syncthreads();
    }
    if (tid == 0) {
        const float n = 1024.f;
        float cm  = red[0][0] / n;
        float cvu = (red[1][0] - n * cm * cm) / (n - 1.f);   // unbiased (adain)
        float cvb = red[1][0] / n - cm * cm;                 // biased (InstanceNorm)
        float cs  = sqrtf(cvu + 1e-5f);
        float sm  = red[2][0] / n;
        float svu = (red[3][0] - n * sm * sm) / (n - 1.f);
        float ssd = sqrtf(svu + 1e-5f);
        float tsc = ssd / cs;
        float tsh = sm - cm * tsc;
        float asc = 0.35355339059327373f * rsqrtf(cvb + 1e-5f); // dn = 64^-0.25 folded in
        float ash = -cm * asc;
        reinterpret_cast<float4*>(stats + (size_t)bc * 4)[0] = make_float4(asc, ash, tsc, tsh);
    }
}

// ---------------------------------------------------------------- K2: performer proj + tok build
__global__ void k_perf(const float* __restrict__ cF, const float* __restrict__ proj,
                       const float* __restrict__ pos_emb, const float* __restrict__ stats,
                       unsigned short* __restrict__ tok) {
    __shared__ __align__(16) unsigned short Alds[256 * 64];
    __shared__ __align__(16) unsigned short Plds[64 * 64];
    int blk = blockIdx.x, tid = threadIdx.x;
    int b = blk >> 5, rem = blk & 31, h = rem >> 1, cg = rem & 1;
    int c0 = cg * 256;

    for (int i = tid; i < 64 * 64; i += 256) Plds[i] = f2bf(proj[i]);  // proj is [j][w] == N-major

    for (int q = tid; q < 256 * 16; q += 256) {
        int row = q >> 4, seg = q & 15, c = c0 + row;
        float4 x = *reinterpret_cast<const float4*>(
            cF + (((size_t)(b * Cc + c)) * Hh + h) * Dd + seg * 4);
        float asc = stats[(size_t)(b * Cc + c) * 4 + 0];
        float ash = stats[(size_t)(b * Cc + c) * 4 + 1];
        unsigned short* dst = &Alds[row * 64 + seg * 4];
        dst[0] = f2bf(x.x * asc + ash); dst[1] = f2bf(x.y * asc + ash);
        dst[2] = f2bf(x.z * asc + ash); dst[3] = f2bf(x.w * asc + ash);
    }
    __syncthreads();

    int wave = tid >> 5, lane = tid & 31, half = lane >> 4;
    for (int mi = 0; mi < 2; ++mi) {
        int mt = wave * 2 + mi;
        // hoist A fragments: loaded once, reused over all 4 N-tiles
        v16bf_t a0 = frag_a(Alds, 64, mt * 16, 0, lane);
        v16bf_t a1 = frag_a(Alds, 64, mt * 16, 32, lane);
        for (int nt = 0; nt < 4; ++nt) {
            v16bf_t b0 = frag_b(Plds, 64, nt * 16, 0, lane);
            v16bf_t b1 = frag_b(Plds, 64, nt * 16, 32, lane);
            v8f_t acc = zero8();
            acc = wmma_bf16(a0, b0, acc);
            acc = wmma_bf16(a1, b1, acc);
            int j = nt * 16 + (lane & 15);
            int t = h * 64 + j;
            int cbase = c0 + mt * 16 + 8 * half;
            float vals[8];
            for (int v = 0; v < 8; ++v) {
                float q = fmaxf(acc[v], 0.f) + 1e-3f;                 // relu + KERNEL_EPS
                vals[v] = q + pos_emb[(size_t)t * Cc + cbase + v];    // + pos emb
            }
            *reinterpret_cast<uint4*>(&tok[((size_t)b * NT + t) * Cc + cbase]) = pack8bf(vals);
        }
    }
}

// ---------------------------------------------------------------- K3: token linear (1024x512x512 per batch)
// One A fragment per k-step feeds 8 register-resident N-tile accumulators.
__global__ void k_linear(const unsigned short* __restrict__ tok,
                         const unsigned short* __restrict__ W_t,
                         const float* __restrict__ b_lin, unsigned short* __restrict__ sp) {
    __shared__ __align__(16) unsigned short Alds[32 * 512];
    int blk = blockIdx.x, tid = threadIdx.x;
    int b = blk >> 5, mg = blk & 31, t0 = mg * 32;

    const uint4* src = reinterpret_cast<const uint4*>(tok + ((size_t)b * NT + t0) * Cc);
    uint4* dst = reinterpret_cast<uint4*>(Alds);
    for (int i = tid; i < 32 * 512 / 8; i += 256) dst[i] = src[i];
    __syncthreads();

    int wave = tid >> 5, lane = tid & 31, half = lane >> 4;
    int mt = wave >> 2;         // 0..1 -> rows [mt*16, mt*16+16)
    int ng = wave & 3;          // 0..3 -> N-tiles [ng*8, ng*8+8)
    v8f_t acc[8];
    for (int i = 0; i < 8; ++i) acc[i] = zero8();

    for (int ks = 0; ks < 512; ks += 32) {
        v16bf_t a = frag_a(Alds, 512, mt * 16, ks, lane);
        if (ks + 32 < 512)      // pull next k-slice of this lane's B rows into cache
            __builtin_prefetch(W_t + (size_t)((ng * 8) * 16 + (lane & 15)) * Cc + ks + 32, 0, 3);
#pragma unroll
        for (int i = 0; i < 8; ++i) {
            int nt = ng * 8 + i;
            v16bf_t bb = frag_b(W_t, 512, nt * 16, ks, lane);
            acc[i] = wmma_bf16(a, bb, acc[i]);
        }
    }
#pragma unroll
    for (int i = 0; i < 8; ++i) {
        int nt = ng * 8 + i;
        int n = nt * 16 + (lane & 15);
        float bl = b_lin[n];
        int tb = t0 + mt * 16 + 8 * half;    // global t for v=0; v consecutive
        int hh = tb >> 6, j0 = tb & 63;
        float vals[8];
        for (int v = 0; v < 8; ++v) vals[v] = acc[i][v] + bl;
        *reinterpret_cast<uint4*>(
            &sp[(((size_t)b * Cc + n) * Hh + hh) * Dd + j0]) = pack8bf(vals);
    }
}

// ---------------------------------------------------------------- K4: VQ argmin + gather (out == qz)
// 32 rows per block: halves L2 traffic on ebt vs 16-row blocks; each B fragment feeds 2 M-tiles.
__global__ void k_vq(const unsigned short* __restrict__ sp, const unsigned short* __restrict__ ebt,
                     const float* __restrict__ halfnorm, const float* __restrict__ embed,
                     float* __restrict__ dout) {
    __shared__ __align__(16) unsigned short Alds[32 * 64];
    __shared__ float redv[8][32];
    __shared__ int   redn[8][32];
    __shared__ int   chosen[32];
    int blk = blockIdx.x, tid = threadIdx.x;
    size_t r0 = (size_t)blk * 32;

    reinterpret_cast<uint4*>(Alds)[tid] = reinterpret_cast<const uint4*>(sp + r0 * 64)[tid];
    __syncthreads();

    int wave = tid >> 5, lane = tid & 31, half = lane >> 4;
    // hoisted A fragments: loaded from LDS exactly once per wave
    v16bf_t a00 = frag_a(Alds, 64, 0,  0,  lane);
    v16bf_t a01 = frag_a(Alds, 64, 0,  32, lane);
    v16bf_t a10 = frag_a(Alds, 64, 16, 0,  lane);
    v16bf_t a11 = frag_a(Alds, 64, 16, 32, lane);

    float best[2][8]; int bidx[2][8];
    for (int m = 0; m < 2; ++m)
        for (int v = 0; v < 8; ++v) { best[m][v] = -3.4e38f; bidx[m][v] = 0; }

    for (int i = 0; i < 8; ++i) {
        int nt = wave * 8 + i;
        v16bf_t b0 = frag_b(ebt, 64, nt * 16, 0,  lane);
        v16bf_t b1 = frag_b(ebt, 64, nt * 16, 32, lane);
        if (i < 7)
            __builtin_prefetch(ebt + (size_t)((nt + 1) * 16 + (lane & 15)) * Dd, 0, 3);
        int n = nt * 16 + (lane & 15);
        float hn = halfnorm[n];

        v8f_t acc0 = zero8();
        acc0 = wmma_bf16(a00, b0, acc0);
        acc0 = wmma_bf16(a01, b1, acc0);
        v8f_t acc1 = zero8();
        acc1 = wmma_bf16(a10, b0, acc1);
        acc1 = wmma_bf16(a11, b1, acc1);

        for (int v = 0; v < 8; ++v) {
            float s0 = acc0[v] - hn;   // argmax of f.e - 0.5||e||^2 == argmin dist
            if (s0 > best[0][v] || (s0 == best[0][v] && n < bidx[0][v])) { best[0][v] = s0; bidx[0][v] = n; }
            float s1 = acc1[v] - hn;
            if (s1 > best[1][v] || (s1 == best[1][v] && n < bidx[1][v])) { best[1][v] = s1; bidx[1][v] = n; }
        }
    }
    // butterfly reduce over the 16 lanes of each half (wave32)
    for (int off = 1; off < 16; off <<= 1) {
        for (int m = 0; m < 2; ++m)
            for (int v = 0; v < 8; ++v) {
                float ov = __shfl_xor(best[m][v], off, 32);
                int   on = __shfl_xor(bidx[m][v], off, 32);
                if (ov > best[m][v] || (ov == best[m][v] && on < bidx[m][v])) {
                    best[m][v] = ov; bidx[m][v] = on;
                }
            }
    }
    if ((lane & 15) == 0) {
        for (int m = 0; m < 2; ++m)
            for (int v = 0; v < 8; ++v) {
                int row = m * 16 + half * 8 + v;
                redv[wave][row] = best[m][v]; redn[wave][row] = bidx[m][v];
            }
    }
    __syncthreads();
    if (tid < 32) {
        float bv = redv[0][tid]; int bn = redn[0][tid];
        for (int w = 1; w < 8; ++w) {
            float v = redv[w][tid]; int n = redn[w][tid];
            if (v > bv || (v == bv && n < bn)) { bv = v; bn = n; }
        }
        chosen[tid] = bn;
        dout[OUT_ELEMS + r0 + tid] = (float)bn;   // ind output
    }
    __syncthreads();
    for (int i = tid; i < 32 * 64; i += 256) {    // out[row][j] = embed[j][ind]
        int row = i >> 6, j = i & 63;
        dout[(r0 + row) * 64 + j] = embed[(size_t)j * NE + chosen[row]];
    }
}

// ---------------------------------------------------------------- K5: per-(b,c) loss partials
__global__ void k_loss_part(const float* __restrict__ dout, const float* __restrict__ cF,
                            const float* __restrict__ stats, float* __restrict__ part) {
    __shared__ float red[5][256];
    int bc = blockIdx.x, tid = threadIdx.x;
    float tsc = stats[(size_t)bc * 4 + 2], tsh = stats[(size_t)bc * 4 + 3];
    float4 q = reinterpret_cast<const float4*>(dout + (size_t)bc * 1024)[tid];
    float4 x = reinterpret_cast<const float4*>(cF + (size_t)bc * 1024)[tid];
    float tq[4] = {q.x, q.y, q.z, q.w};
    float tt[4] = {x.x * tsc + tsh, x.y * tsc + tsh, x.z * tsc + tsh, x.w * tsc + tsh};
    float d2 = 0, sq = 0, sq2 = 0, st = 0, st2 = 0;
    for (int k = 0; k < 4; ++k) {
        float d = tq[k] - tt[k];
        d2 += d * d; sq += tq[k]; sq2 += tq[k] * tq[k]; st += tt[k]; st2 += tt[k] * tt[k];
    }
    red[0][tid] = d2; red[1][tid] = sq; red[2][tid] = sq2; red[3][tid] = st; red[4][tid] = st2;
    __syncthreads();
    for (int o = 128; o > 0; o >>= 1) {
        if (tid < o)
            for (int j = 0; j < 5; ++j) red[j][tid] += red[j][tid + o];
        __syncthreads();
    }
    if (tid == 0) {
        const float n = 1024.f;
        float pm = red[1][0] / n, tm = red[3][0] / n;
        float ps = sqrtf((red[2][0] - n * pm * pm) / (n - 1.f) + 1e-5f);
        float ts = sqrtf((red[4][0] - n * tm * tm) / (n - 1.f) + 1e-5f);
        part[(size_t)bc * 3 + 0] = red[0][0];
        part[(size_t)bc * 3 + 1] = (pm - tm) * (pm - tm);
        part[(size_t)bc * 3 + 2] = (ps - ts) * (ps - ts);
    }
}

// ---------------------------------------------------------------- K6: final scalar
__global__ void k_loss_final(const float* __restrict__ part, float* __restrict__ dout) {
    __shared__ float red[3][256];
    int tid = threadIdx.x;
    float a = 0, b = 0, c = 0;
    for (int i = tid; i < BC; i += 256) {
        a += part[(size_t)i * 3 + 0]; b += part[(size_t)i * 3 + 1]; c += part[(size_t)i * 3 + 2];
    }
    red[0][tid] = a; red[1][tid] = b; red[2][tid] = c;
    __syncthreads();
    for (int o = 128; o > 0; o >>= 1) {
        if (tid < o)
            for (int j = 0; j < 3; ++j) red[j][tid] += red[j][tid + o];
        __syncthreads();
    }
    if (tid == 0)
        dout[OUT_ELEMS + NROWS] =
            red[0][0] / (float)OUT_ELEMS + 5.f * ((red[1][0] + red[2][0]) / (float)BC);
}

// ---------------------------------------------------------------- launch
extern "C" void kernel_launch(void* const* d_in, const int* in_sizes, int n_in,
                              void* d_out, int out_size, void* d_ws, size_t ws_size,
                              hipStream_t stream) {
    const float* cF      = (const float*)d_in[0];
    const float* sF      = (const float*)d_in[1];
    const float* embed   = (const float*)d_in[2];
    const float* pos_emb = (const float*)d_in[3];
    const float* W_lin   = (const float*)d_in[4];
    const float* b_lin   = (const float*)d_in[5];
    const float* proj    = (const float*)d_in[6];
    float* dout = (float*)d_out;

    char* ws = (char*)d_ws;
    float*          stats    = (float*)(ws + 0);                      // 8192*4 f32  = 128 KB
    unsigned short* W_t      = (unsigned short*)(ws + 131072);        // 512*512 bf16 = 512 KB
    unsigned short* ebt      = (unsigned short*)(ws + 655360);        // 1024*64 bf16 = 128 KB
    float*          halfnorm = (float*)(ws + 786432);                 // 1024 f32 = 4 KB
    unsigned short* tok      = (unsigned short*)(ws + 790528);        // 16 MB
    unsigned short* sp       = (unsigned short*)(ws + 790528 + 16777216); // 16 MB
    float*          part     = (float*)(ws + 790528 + 2 * 16777216);  // 8192*3 f32

    k_prep      <<<1024, 256, 0, stream>>>(W_lin, embed, W_t, ebt, halfnorm);
    k_stats     <<<BC,   256, 0, stream>>>(cF, sF, stats);
    k_perf      <<<Bn * Hh * 2, 256, 0, stream>>>(cF, proj, pos_emb, stats, tok);
    k_linear    <<<Bn * 32,     256, 0, stream>>>(tok, W_t, b_lin, sp);
    k_vq        <<<NROWS / 32,  256, 0, stream>>>(sp, ebt, halfnorm, embed, dout);
    k_loss_part <<<BC,   256, 0, stream>>>(dout, cF, stats, part);
    k_loss_final<<<1,    256, 0, stream>>>(part, dout);
}
